// Simple_Self_Attention_68564857914107
// MI455X (gfx1250) — matrix operand order repaired and checked by reading, not compiled, not run on hardware
//
#include <hip/hip_runtime.h>

// ---------------------------------------------------------------------------
// Simple self-attention, CDNA5 (gfx1250, wave32) WMMA bf16 pipeline.
//   B=4, S=2048, D=U=1024. ~120 GFLOP, compute-bound -> bf16 WMMA everywhere.
//   v2: 64x64 tiles/wave in projections (1 b128 per WMMA), 256-wide K blocks
//       in flash attention (4x fewer barriers, A-fragment reuse), LDS-tiled
//       weight transpose for coalesced conversion.
// ---------------------------------------------------------------------------

typedef __attribute__((ext_vector_type(16))) __bf16 v16bf;
typedef __attribute__((ext_vector_type(8)))  __bf16 v8bf;
typedef __attribute__((ext_vector_type(8)))  float  v8f;

#define B_  4
#define S_  2048
#define D_  1024
#define U_  1024
#define M_  (B_ * S_)        // 8192 rows for the projection GEMMs

__device__ __forceinline__ unsigned short f32_to_bf16_rte(float f) {
    union { float f; unsigned u; } v; v.f = f;
    unsigned r = v.u + 0x7FFFu + ((v.u >> 16) & 1u);   // round-to-nearest-even
    return (unsigned short)(r >> 16);
}

union FragAB { v16bf v; struct { v8bf lo, hi; } h; };

// A-fragment (16x32 bf16): lane l -> row = l%16; elems 0..7 at K=k0+(l/16)*8,
// elems 8..15 at K=k0+16+(l/16)*8. Two 16B loads.
__device__ __forceinline__ v16bf load_frag_a(const __bf16* row, int k0, int hl) {
    FragAB f;
    f.h.lo = *(const v8bf*)(row + k0 + hl * 8);
    f.h.hi = *(const v8bf*)(row + k0 + 16 + hl * 8);
    return f.v;
}

// B-fragment (32x16 bf16): lane l -> col = l%16; elems 0..7 at K=k0+(l/16)*8,
// elems 8..15 at K=k0+16+(l/16)*8.
// `row` points at the (column-owning) row of a pre-transposed matrix. Two 16B loads.
__device__ __forceinline__ v16bf load_frag_b(const __bf16* row, int k0, int hl) {
    FragAB f;
    f.h.lo = *(const v8bf*)(row + k0 + hl * 8);
    f.h.hi = *(const v8bf*)(row + k0 + 16 + hl * 8);
    return f.v;
}

__device__ __forceinline__ v8f wmma_bf16(v16bf a, v16bf b, v8f c) {
    return __builtin_amdgcn_wmma_f32_16x16x32_bf16(false, a, false, b,
                                                   (short)0, c, false, false);
}

// ---------------------------------------------------------------------------
// f32 -> bf16 elementwise
__global__ __launch_bounds__(256) void cvt_bf16_kernel(const float* __restrict__ src,
                                                       unsigned short* __restrict__ dst,
                                                       int n) {
    int i = blockIdx.x * blockDim.x + threadIdx.x;
    if (i < n) dst[i] = f32_to_bf16_rte(src[i]);
}

// W [D,U] f32 -> WT [U,D] bf16, 32x32 LDS tile so both sides are coalesced.
__global__ __launch_bounds__(256) void cvt_wT_kernel(const float* __restrict__ W,
                                                     unsigned short* __restrict__ WT) {
    __shared__ unsigned short t[32][33];
    const int td0 = (blockIdx.x & (D_ / 32 - 1)) * 32;   // d tile origin
    const int tu0 = (blockIdx.x / (D_ / 32)) * 32;       // u tile origin
    const int c  = threadIdx.x & 31;
    const int rb = (threadIdx.x >> 5) * 4;
#pragma unroll
    for (int i = 0; i < 4; ++i) {
        const int d = td0 + rb + i;
        const int u = tu0 + c;
        t[c][rb + i] = f32_to_bf16_rte(W[(size_t)d * U_ + u]);  // t[u_loc][d_loc]
    }
    __syncthreads();
#pragma unroll
    for (int i = 0; i < 4; ++i) {
        const int u = tu0 + rb + i;
        const int d = td0 + c;
        WT[(size_t)u * D_ + d] = t[rb + i][c];
    }
}

// ---------------------------------------------------------------------------
// Projection GEMM: out[m,n] = sum_k X[m,k]*WT[n,k] + bias[n], bf16 in/out.
// One wave computes a 64x64 tile as a 4x4 grid of 16x16 WMMA fragments:
// per 32-wide k-step, 8 fragment loads feed 16 WMMAs (1 b128 per WMMA).
// transposed==1 stores V as [B][U][S] (per-batch transposed) for P*V.
__global__ __launch_bounds__(128) void proj_gemm_kernel(const __bf16* __restrict__ X,
                                                        const __bf16* __restrict__ WT,
                                                        const float* __restrict__ bias,
                                                        unsigned short* __restrict__ out,
                                                        int transposed) {
    const int lane = threadIdx.x & 31;
    const int wid  = threadIdx.x >> 5;
    const int hl   = lane >> 4;
    const int lc   = lane & 15;

    const int tile = blockIdx.x * 4 + wid;     // (M/64)*(U/64) = 2048 tiles
    const int mt   = tile >> 4;                // 128 M-tiles of 64
    const int nt   = tile & 15;                // 16  N-tiles of 64
    const int m0   = mt * 64;
    const int n0   = nt * 64;

    const __bf16* arow[4];
    const __bf16* brow[4];
#pragma unroll
    for (int i = 0; i < 4; ++i) {
        arow[i] = X  + (size_t)(m0 + i * 16 + lc) * D_;
        brow[i] = WT + (size_t)(n0 + i * 16 + lc) * D_;
    }

    v8f c[4][4] = {};
#pragma unroll 1
    for (int k = 0; k < D_; k += 32) {
        v16bf a[4], b[4];
#pragma unroll
        for (int i = 0; i < 4; ++i) a[i] = load_frag_a(arow[i], k, hl);
#pragma unroll
        for (int j = 0; j < 4; ++j) b[j] = load_frag_b(brow[j], k, hl);
#pragma unroll
        for (int i = 0; i < 4; ++i)
#pragma unroll
            for (int j = 0; j < 4; ++j)
                c[i][j] = wmma_bf16(a[i], b[j], c[i][j]);
    }

    float bv[4];
#pragma unroll
    for (int j = 0; j < 4; ++j) bv[j] = bias[n0 + j * 16 + lc];

#pragma unroll
    for (int i = 0; i < 4; ++i) {
#pragma unroll
        for (int r = 0; r < 8; ++r) {
            const int m = m0 + i * 16 + r + hl * 8;
#pragma unroll
            for (int j = 0; j < 4; ++j) {
                const int n = n0 + j * 16 + lc;
                unsigned short v = f32_to_bf16_rte(c[i][j][r] + bv[j]);
                if (!transposed) {
                    out[(size_t)m * U_ + n] = v;
                } else {
                    const int bb = m >> 11;        // batch (S_=2048)
                    const int tt = m & (S_ - 1);
                    out[((size_t)bb * U_ + n) * S_ + tt] = v;
                }
            }
        }
    }
}

// ---------------------------------------------------------------------------
// Flash attention: grid = B * (S/16) blocks of 128 threads (4 waves).
// Block owns 16 query rows. Per iteration the block consumes a 256-wide K/V
// block: wave w computes a 16x64 score strip (A-fragment reused over 4 WMMAs),
// online softmax runs over the 16x256 strip in LDS, then each wave accumulates
// its 256 U-columns of O += P*V (8 k-steps x 16 N-tiles of WMMA).
__global__ __launch_bounds__(128) void attn_kernel(const __bf16* __restrict__ Qb,
                                                   const __bf16* __restrict__ Kb,
                                                   const __bf16* __restrict__ VT,
                                                   float* __restrict__ out) {
    const int tid  = threadIdx.x;
    const int lane = tid & 31;
    const int wid  = tid >> 5;
    const int hl   = lane >> 4;
    const int lc   = lane & 15;
    const int bidx = blockIdx.x >> 7;      // batch
    const int q0   = (blockIdx.x & 127) * 16;

    __shared__ __align__(16) float          Sf[16][260];   // +4 pad: bank stagger
    __shared__ __align__(16) unsigned short Pl[16][264];   // +8 pad keeps 16B align
    __shared__ float part[16][8];
    __shared__ float mrow[16], lrow[16], arow_s[16];

    if (tid < 16) { mrow[tid] = -1e30f; lrow[tid] = 0.f; }
    __syncthreads();

    const __bf16* qrow = Qb + (size_t)(bidx * S_ + q0 + lc) * U_;

    v8f O[16] = {};                        // 16 rows x 256 cols per wave
    const float scale = 0.03125f;          // 1/sqrt(1024)
    const int r = tid & 15, g = tid >> 4;  // softmax indexing: 8 groups x 32 cols

    for (int jb = 0; jb < S_ / 256; ++jb) {
        // ---- scores: this wave's 16x64 strip of S = Q K^T ----
        const int t0 = jb * 256 + wid * 64;
        const __bf16* krow[4];
#pragma unroll
        for (int j = 0; j < 4; ++j)
            krow[j] = Kb + (size_t)(bidx * S_ + t0 + j * 16 + lc) * U_;

        v8f s[4] = {};
#pragma unroll 1
        for (int k = 0; k < U_; k += 32) {
            v16bf a = load_frag_a(qrow, k, hl);
#pragma unroll
            for (int j = 0; j < 4; ++j)
                s[j] = wmma_bf16(a, load_frag_b(krow[j], k, hl), s[j]);
        }
#pragma unroll
        for (int j = 0; j < 4; ++j)
#pragma unroll
            for (int rr = 0; rr < 8; ++rr)
                Sf[rr + hl * 8][wid * 64 + j * 16 + lc] = s[j][rr] * scale;
        __syncthreads();

        // ---- online softmax over the 16x256 strip ----
        float pm = -1e30f;
#pragma unroll 4
        for (int c = 0; c < 32; ++c) pm = fmaxf(pm, Sf[r][g * 32 + c]);
        part[r][g] = pm;
        __syncthreads();
        if (tid < 16) {
            float tm = mrow[tid];
#pragma unroll
            for (int gg = 0; gg < 8; ++gg) tm = fmaxf(tm, part[tid][gg]);
            arow_s[tid] = __expf(mrow[tid] - tm);   // 0 on first block
            mrow[tid]   = tm;
        }
        __syncthreads();
        {
            const float mr = mrow[r];
            float ps = 0.f;
#pragma unroll 4
            for (int c = 0; c < 32; ++c) {
                float p = __expf(Sf[r][g * 32 + c] - mr);
                ps += p;
                Pl[r][g * 32 + c] = f32_to_bf16_rte(p);
            }
            part[r][g] = ps;
        }
        __syncthreads();
        if (tid < 16) {
            float s2 = 0.f;
#pragma unroll
            for (int gg = 0; gg < 8; ++gg) s2 += part[tid][gg];
            lrow[tid] = lrow[tid] * arow_s[tid] + s2;
        }
        __syncthreads();

        // ---- rescale accumulators by per-row alpha ----
        float al[8];
#pragma unroll
        for (int rr = 0; rr < 8; ++rr) al[rr] = arow_s[rr + hl * 8];
#pragma unroll
        for (int nt = 0; nt < 16; ++nt) {
#pragma unroll
            for (int rr = 0; rr < 8; ++rr) O[nt][rr] *= al[rr];
        }

        // ---- O += P (16x256) x V (256 x 256-per-wave) ----
        const __bf16* prow = (const __bf16*)&Pl[lc][0];
#pragma unroll 1
        for (int kk = 0; kk < 8; ++kk) {
            v16bf aP = load_frag_a(prow, kk * 32, hl);
#pragma unroll
            for (int nt = 0; nt < 16; ++nt) {
                const __bf16* vrow =
                    VT + (size_t)(bidx * U_ + wid * 256 + nt * 16 + lc) * S_;
                v16bf b = load_frag_b(vrow, jb * 256 + kk * 32, hl);
                O[nt] = wmma_bf16(aP, b, O[nt]);
            }
        }
        __syncthreads();
    }

    // ---- normalize and write f32 output ----
    float inv[8];
#pragma unroll
    for (int rr = 0; rr < 8; ++rr) inv[rr] = 1.f / lrow[rr + hl * 8];
#pragma unroll
    for (int nt = 0; nt < 16; ++nt) {
#pragma unroll
        for (int rr = 0; rr < 8; ++rr) {
            const int row = q0 + rr + hl * 8;
            const int col = wid * 256 + nt * 16 + lc;
            out[(size_t)(bidx * S_ + row) * U_ + col] = O[nt][rr] * inv[rr];
        }
    }
}

// ---------------------------------------------------------------------------
extern "C" void kernel_launch(void* const* d_in, const int* in_sizes, int n_in,
                              void* d_out, int out_size, void* d_ws, size_t ws_size,
                              hipStream_t stream) {
    const float* query = (const float*)d_in[0];
    const float* value = (const float*)d_in[1];
    const float* Wq    = (const float*)d_in[2];
    const float* bq    = (const float*)d_in[3];
    const float* Wk    = (const float*)d_in[4];
    const float* bk    = (const float*)d_in[5];
    const float* Wv    = (const float*)d_in[6];
    const float* bv    = (const float*)d_in[7];
    float* out = (float*)d_out;

    // Workspace layout (bf16 buffers), total ~86 MiB.
    char* ws = (char*)d_ws;
    const size_t szX = (size_t)M_ * D_ * 2;   // 16 MiB each
    const size_t szW = (size_t)D_ * U_ * 2;   // 2 MiB each
    unsigned short* Xq  = (unsigned short*)(ws);                     // [M,D]
    unsigned short* Xv  = (unsigned short*)(ws + szX);               // [M,D]
    unsigned short* WqT = (unsigned short*)(ws + 2 * szX);           // [U,D]
    unsigned short* WkT = (unsigned short*)(ws + 2 * szX + szW);
    unsigned short* WvT = (unsigned short*)(ws + 2 * szX + 2 * szW);
    unsigned short* Qb  = (unsigned short*)(ws + 2 * szX + 3 * szW);            // [M,U]
    unsigned short* Kb  = (unsigned short*)(ws + 3 * szX + 3 * szW);            // [M,U]
    unsigned short* VTb = (unsigned short*)(ws + 4 * szX + 3 * szW);            // [B,U,S]

    // 1) convert activations + transpose-convert weights to bf16
    const int nX = M_ * D_;
    cvt_bf16_kernel<<<nX / 256, 256, 0, stream>>>(query, Xq, nX);
    cvt_bf16_kernel<<<nX / 256, 256, 0, stream>>>(value, Xv, nX);
    const int nWblk = (D_ / 32) * (U_ / 32);   // 1024 tile blocks
    cvt_wT_kernel<<<nWblk, 256, 0, stream>>>(Wq, WqT);
    cvt_wT_kernel<<<nWblk, 256, 0, stream>>>(Wk, WkT);
    cvt_wT_kernel<<<nWblk, 256, 0, stream>>>(Wv, WvT);

    // 2) projections: Q, K row-major; V transposed per batch for P*V
    const int projBlocks = (M_ / 64) * (U_ / 64) / 4;   // 512 blocks x 4 waves
    proj_gemm_kernel<<<projBlocks, 128, 0, stream>>>((const __bf16*)Xq, (const __bf16*)WqT, bq, Qb, 0);
    proj_gemm_kernel<<<projBlocks, 128, 0, stream>>>((const __bf16*)Xv, (const __bf16*)WkT, bk, Kb, 0);
    proj_gemm_kernel<<<projBlocks, 128, 0, stream>>>((const __bf16*)Xv, (const __bf16*)WvT, bv, VTb, 1);

    // 3) flash attention
    attn_kernel<<<B_ * (S_ / 16), 128, 0, stream>>>((const __bf16*)Qb, (const __bf16*)Kb,
                                                    (const __bf16*)VTb, out);
}
